// OSS_27513560498491
// MI455X (gfx1250) — matrix operand order repaired and checked by reading, not compiled, not run on hardware
//
#include <hip/hip_runtime.h>
#include <hip/hip_bf16.h>
#include <cstdint>

typedef __attribute__((ext_vector_type(16))) _Float16 v16h;
typedef __attribute__((ext_vector_type(8)))  _Float16 v8h;
typedef __attribute__((ext_vector_type(2)))  _Float16 v2h;
typedef __attribute__((ext_vector_type(8)))  float    v8f;

#define DEV __device__ __forceinline__

constexpr int Bn = 4, Cn = 32, HW = 4096;
constexpr int M1 = Bn * HW;           // 16384 rows for mamba1
constexpr int DIN = 256;              // d_in (mamba1)
constexpr int DMODEL = 128;           // d_model (mamba1)
constexpr int NCH = 8;                // scan chunks along L
constexpr int CHL = HW / NCH;         // 512

DEV float siluf(float v)     { return v / (1.f + __expf(-v)); }
DEV float softplusf(float v) { return v > 20.f ? v : log1pf(__expf(v)); }

// ---------------------------------------------------------------- prep: fop (B,L,128) f16
// feature k = g*32 + c, g in {h_fwd, h_bwd, w_fwd, w_bwd}
__global__ void k_prep_fop(const float* __restrict__ fo1, _Float16* __restrict__ fopH) {
  int idx = blockIdx.x * 256 + threadIdx.x;
  if (idx >= M1 * DMODEL) return;
  int k = idx & (DMODEL - 1);
  int m = idx >> 7;
  int b = m >> 12, l = m & (HW - 1);
  int g = k >> 5, c = k & 31;
  int lr  = HW - 1 - l;
  int lw  = (l  & 63) * 64 + (l  >> 6);
  int lwr = (lr & 63) * 64 + (lr >> 6);
  int sp = (g == 0) ? l : (g == 1) ? lr : (g == 2) ? lw : lwr;
  fopH[idx] = (_Float16)fo1[((size_t)(b * Cn + c) << 12) + sp];
}

// ---------------------------------------------------------------- prep: weights -> f16 (x_w padded 40->48 rows)
__global__ void k_prep_w(const float* __restrict__ inw, const float* __restrict__ xw,
                         const float* __restrict__ outw,
                         _Float16* __restrict__ inwH, _Float16* __restrict__ xwH,
                         _Float16* __restrict__ outwH) {
  int idx = blockIdx.x * 256 + threadIdx.x;
  const int n1 = 512 * 128, n2 = 48 * 256, n3 = 128 * 256;
  if (idx < n1) { inwH[idx] = (_Float16)inw[idx]; return; }
  idx -= n1;
  if (idx < n2) {
    int r = idx >> 8;
    xwH[idx] = (r < 40) ? (_Float16)xw[idx] : (_Float16)0.f;
    return;
  }
  idx -= n2;
  if (idx < n3) outwH[idx] = (_Float16)outw[idx];
}

// ---------------------------------------------------------------- WMMA fragment loaders (ISA VGPR layouts)
// A 16x32 f16: lane m=lane&15, g=lane>>4; element e -> K = (e&8?16:0) + g*8 + (e&7)
DEV v16h load_fragA(const _Float16* __restrict__ rowp, int g) {
  v16h r;
#pragma unroll
  for (int j = 0; j < 8; ++j) {
    int koff = ((j & 4) << 2) + g * 8 + ((j & 3) << 1);   // pairs of consecutive K
    v2h p = *(const v2h*)(rowp + koff);
    r[2 * j] = p[0]; r[2 * j + 1] = p[1];
  }
  return r;
}
// B 32x16 f16: lane n=lane&15, g=lane>>4; element e -> K = (e&8?16:0) + g*8 + (e&7)
DEV v16h load_fragB(const _Float16* __restrict__ rowp, int g) {
  v8h lo = *(const v8h*)(rowp + g * 8);
  v8h hi = *(const v8h*)(rowp + 16 + g * 8);
  v16h r;
#pragma unroll
  for (int j = 0; j < 8; ++j) { r[j] = lo[j]; r[8 + j] = hi[j]; }
  return r;
}

// ---------------------------------------------------------------- generic WMMA GEMM: C(MxN) = A(MxK) * W(NxK)^T
template <int K>
__global__ void k_gemm(const _Float16* __restrict__ A, const _Float16* __restrict__ Bw,
                       float* __restrict__ C, int N) {
  int wave = threadIdx.x >> 5, lane = threadIdx.x & 31;
  int row = lane & 15, g = lane >> 4;
  int mt = blockIdx.x;
  int nt = blockIdx.y * (blockDim.x >> 5) + wave;
  if (nt * 16 >= N) return;
  const _Float16* Arow = A + (size_t)(mt * 16 + row) * K;
  const _Float16* Brow = Bw + (size_t)(nt * 16 + row) * K;
  v8f acc = {0.f, 0.f, 0.f, 0.f, 0.f, 0.f, 0.f, 0.f};
#pragma unroll
  for (int k0 = 0; k0 < K; k0 += 32) {
    v16h a = load_fragA(Arow + k0, g);
    v16h b = load_fragB(Brow + k0, g);
    acc = __builtin_amdgcn_wmma_f32_16x16x32_f16(false, a, false, b, (short)0, acc,
                                                 false, false);
  }
  int cm = mt * 16 + g * 8, cn = nt * 16 + row;   // D layout: VGPR v -> M=g*8+v, N=lane&15
#pragma unroll
  for (int v = 0; v < 8; ++v) C[(size_t)(cm + v) * N + cn] = acc[v];
}

// ---------------------------------------------------------------- causal depthwise conv(4) + SiLU
__global__ void k_conv(const float* __restrict__ xz, const float* __restrict__ cw,
                       const float* __restrict__ cb, float* __restrict__ xcF,
                       _Float16* __restrict__ xcH) {
  int idx = blockIdx.x * 256 + threadIdx.x;
  if (idx >= M1 * DIN) return;
  int d = idx & 255, m = idx >> 8;
  int l = m & (HW - 1);
  float acc = cb[d];
#pragma unroll
  for (int j = 0; j < 4; ++j) {
    int ll = l - 3 + j;
    if (ll >= 0) acc += cw[d * 4 + j] * xz[((size_t)(m - l + ll) << 9) + d];
  }
  float s = siluf(acc);
  xcF[idx] = s;
  xcH[idx] = (_Float16)s;
}

// ---------------------------------------------------------------- selective scan, pass 1: per-chunk local state + decay product
__global__ void k_scan_part(const float* __restrict__ dbl, const float* __restrict__ xc,
                            const float* __restrict__ dtw_g, const float* __restrict__ dtb_g,
                            const float* __restrict__ Alog,
                            float* __restrict__ hloc, float* __restrict__ Pb) {
  int blk = blockIdx.x;                 // b*64 + dg*8 + ch
  int ch = blk & (NCH - 1);
  int dg = (blk >> 3) & 7;
  int b  = blk >> 6;
  int lane = threadIdx.x & 31;
  int d = dg * 32 + lane;
  float dtw[8];
#pragma unroll
  for (int r = 0; r < 8; ++r) dtw[r] = dtw_g[d * 8 + r];
  float dtb = dtb_g[d];
  float Ad[16];
#pragma unroll
  for (int n = 0; n < 16; ++n) Ad[n] = -__expf(Alog[d * 16 + n]);
  float h[16], P[16];
#pragma unroll
  for (int n = 0; n < 16; ++n) { h[n] = 0.f; P[n] = 1.f; }
  size_t mbase = (size_t)b * HW;
  int l0 = ch * CHL;
  for (int l = l0; l < l0 + CHL; ++l) {
    const float* rowp = dbl + (mbase + l) * 48;
    float rv[40];
#pragma unroll
    for (int q = 0; q < 10; ++q) {
      float4 t = ((const float4*)rowp)[q];
      rv[4 * q] = t.x; rv[4 * q + 1] = t.y; rv[4 * q + 2] = t.z; rv[4 * q + 3] = t.w;
    }
    float s = dtb;
#pragma unroll
    for (int r = 0; r < 8; ++r) s += rv[r] * dtw[r];
    float dtv = softplusf(s);
    float x = xc[((mbase + l) << 8) + d];
    float dx = dtv * x;
#pragma unroll
    for (int n = 0; n < 16; ++n) {
      float dA = __expf(dtv * Ad[n]);
      h[n] = dA * h[n] + dx * rv[8 + n];
      P[n] *= dA;
    }
  }
  size_t o = ((((size_t)b * NCH + ch) * 256) + d) * 16;
#pragma unroll
  for (int n = 0; n < 16; ++n) { hloc[o + n] = h[n]; Pb[o + n] = P[n]; }
}

// ---------------------------------------------------------------- pass 2: sequential carry fix-up (overwrites hloc with h_in)
__global__ void k_scan_fix(float* __restrict__ hloc, const float* __restrict__ Pb) {
  int blk = blockIdx.x;                 // b*8 + dg
  int dg = blk & 7, b = blk >> 3;
  int d = dg * 32 + (threadIdx.x & 31);
  float h[16];
#pragma unroll
  for (int n = 0; n < 16; ++n) h[n] = 0.f;
  for (int c = 0; c < NCH; ++c) {
    size_t o = ((((size_t)b * NCH + c) * 256) + d) * 16;
#pragma unroll
    for (int n = 0; n < 16; ++n) {
      float hl = hloc[o + n], p = Pb[o + n];
      hloc[o + n] = h[n];               // store h_in for this chunk
      h[n] = p * h[n] + hl;             // advance carry
    }
  }
}

// ---------------------------------------------------------------- pass 3: recompute with correct h_in, emit gated y (f16)
__global__ void k_scan_emit(const float* __restrict__ dbl, const float* __restrict__ xc,
                            const float* __restrict__ xz,
                            const float* __restrict__ dtw_g, const float* __restrict__ dtb_g,
                            const float* __restrict__ Alog, const float* __restrict__ Dp,
                            const float* __restrict__ hin, _Float16* __restrict__ yactH) {
  int blk = blockIdx.x;
  int ch = blk & (NCH - 1);
  int dg = (blk >> 3) & 7;
  int b  = blk >> 6;
  int lane = threadIdx.x & 31;
  int d = dg * 32 + lane;
  float dtw[8];
#pragma unroll
  for (int r = 0; r < 8; ++r) dtw[r] = dtw_g[d * 8 + r];
  float dtb = dtb_g[d];
  float DD  = Dp[d];
  float Ad[16];
#pragma unroll
  for (int n = 0; n < 16; ++n) Ad[n] = -__expf(Alog[d * 16 + n]);
  float h[16];
  size_t o = ((((size_t)b * NCH + ch) * 256) + d) * 16;
#pragma unroll
  for (int n = 0; n < 16; ++n) h[n] = hin[o + n];
  size_t mbase = (size_t)b * HW;
  int l0 = ch * CHL;
  for (int l = l0; l < l0 + CHL; ++l) {
    const float* rowp = dbl + (mbase + l) * 48;
    float rv[40];
#pragma unroll
    for (int q = 0; q < 10; ++q) {
      float4 t = ((const float4*)rowp)[q];
      rv[4 * q] = t.x; rv[4 * q + 1] = t.y; rv[4 * q + 2] = t.z; rv[4 * q + 3] = t.w;
    }
    float s = dtb;
#pragma unroll
    for (int r = 0; r < 8; ++r) s += rv[r] * dtw[r];
    float dtv = softplusf(s);
    float x = xc[((mbase + l) << 8) + d];
    float dx = dtv * x;
    float y = 0.f;
#pragma unroll
    for (int n = 0; n < 16; ++n) {
      float dA = __expf(dtv * Ad[n]);
      h[n] = dA * h[n] + dx * rv[8 + n];
      y += h[n] * rv[24 + n];
    }
    y += x * DD;
    float z = xz[((mbase + l) << 9) + 256 + d];
    y *= siluf(z);
    yactH[((mbase + l) << 8) + d] = (_Float16)y;
  }
}

// ---------------------------------------------------------------- 4-direction sum * fo2, channel means
__global__ void k_resid(const float* __restrict__ out1, const float* __restrict__ fo2,
                        float* __restrict__ resid, float* __restrict__ foc) {
  __shared__ float red[256];
  int blk = blockIdx.x;                 // b*32 + c
  int c = blk & 31, b = blk >> 5;
  int tid = threadIdx.x;
  float acc = 0.f;
  for (int p = tid; p < HW; p += 256) {
    const float* r = out1 + ((size_t)(b * HW + p) << 7);
    float v = r[c] + r[32 + c] + r[64 + c] + r[96 + c];
    v *= fo2[((size_t)blk << 12) + p];
    resid[((size_t)blk << 12) + p] = v;
    acc += v;
  }
  red[tid] = acc;
  __syncthreads();
  for (int s = 128; s > 0; s >>= 1) {
    if (tid < s) red[tid] += red[tid + s];
    __syncthreads();
  }
  if (tid == 0) foc[blk] = red[0] * (1.f / 4096.f);
}

// ---------------------------------------------------------------- tiny channel mamba: one wave per batch
__global__ void k_mamba2(const float* __restrict__ foc,
                         const float* __restrict__ inw, const float* __restrict__ cw,
                         const float* __restrict__ cb,  const float* __restrict__ xw,
                         const float* __restrict__ dtw, const float* __restrict__ dtb,
                         const float* __restrict__ Alog, const float* __restrict__ Dp,
                         const float* __restrict__ outw, float* __restrict__ sbuf) {
  __shared__ float xpre[32][4], xc2[32][4], dt2[32][4], Bm2[32][16], Cm2[32][16], y2[32][4];
  int b = blockIdx.x;
  int c = threadIdx.x;                  // 0..31 == sequence position
  float u0 = foc[b * 32 + c];
  float u1 = foc[b * 32 + 31 - c];
  float zl[4];
#pragma unroll
  for (int j = 0; j < 4; ++j) {
    xpre[c][j] = inw[j * 2] * u0 + inw[j * 2 + 1] * u1;
    zl[j]      = inw[(4 + j) * 2] * u0 + inw[(4 + j) * 2 + 1] * u1;
  }
  __syncthreads();
  float xcl[4];
#pragma unroll
  for (int d = 0; d < 4; ++d) {
    float a = cb[d];
#pragma unroll
    for (int j = 0; j < 4; ++j) {
      int cc = c - 3 + j;
      if (cc >= 0) a += cw[d * 4 + j] * xpre[cc][d];
    }
    xcl[d] = siluf(a);
    xc2[c][d] = xcl[d];
  }
  float d0 = 0.f;
#pragma unroll
  for (int d = 0; d < 4; ++d) d0 += xcl[d] * xw[d];          // dt_rank = 1
#pragma unroll
  for (int d = 0; d < 4; ++d) dt2[c][d] = softplusf(d0 * dtw[d] + dtb[d]);
#pragma unroll
  for (int n = 0; n < 16; ++n) {
    float bb = 0.f, cc = 0.f;
#pragma unroll
    for (int d = 0; d < 4; ++d) {
      bb += xcl[d] * xw[(1 + n) * 4 + d];
      cc += xcl[d] * xw[(17 + n) * 4 + d];
    }
    Bm2[c][n] = bb; Cm2[c][n] = cc;
  }
  __syncthreads();
  if (c < 4) {                          // lanes 0..3 run the 32-step scan, one channel each
    int d = c;
    float A2[16], h[16];
#pragma unroll
    for (int n = 0; n < 16; ++n) { A2[n] = -__expf(Alog[d * 16 + n]); h[n] = 0.f; }
    for (int t = 0; t < 32; ++t) {
      float dtv = dt2[t][d];
      float dx = dtv * xc2[t][d];
      float y = 0.f;
#pragma unroll
      for (int n = 0; n < 16; ++n) {
        float dA = __expf(dtv * A2[n]);
        h[n] = dA * h[n] + dx * Bm2[t][n];
        y += h[n] * Cm2[t][n];
      }
      y2[t][d] = y;
    }
  }
  __syncthreads();
  float s = 0.f;
#pragma unroll
  for (int d = 0; d < 4; ++d) {
    float yv = y2[c][d] + xc2[c][d] * Dp[d];
    yv *= siluf(zl[d]);
    s += yv * (outw[d] + outw[4 + d]);  // out@out_w.T then sum over d_model=2
  }
  sbuf[b * 32 + c] = s;
}

// ---------------------------------------------------------------- final: residual * (1 + s[b,c])
__global__ void k_final(const float* __restrict__ resid, const float* __restrict__ sbuf,
                        float* __restrict__ out) {
  int idx = blockIdx.x * 256 + threadIdx.x;
  if (idx >= Bn * Cn * HW) return;
  out[idx] = resid[idx] * (1.f + sbuf[idx >> 12]);
}

// ================================================================ host
extern "C" void kernel_launch(void* const* d_in, const int* in_sizes, int n_in,
                              void* d_out, int out_size, void* d_ws, size_t ws_size,
                              hipStream_t stream) {
  (void)in_sizes; (void)n_in; (void)out_size; (void)ws_size;
  const float* fo1       = (const float*)d_in[0];
  const float* fo2       = (const float*)d_in[1];
  const float* m1_in_w   = (const float*)d_in[2];
  const float* m1_conv_w = (const float*)d_in[3];
  const float* m1_conv_b = (const float*)d_in[4];
  const float* m1_x_w    = (const float*)d_in[5];
  const float* m1_dt_w   = (const float*)d_in[6];
  const float* m1_dt_b   = (const float*)d_in[7];
  const float* m1_A_log  = (const float*)d_in[8];
  const float* m1_D      = (const float*)d_in[9];
  const float* m1_out_w  = (const float*)d_in[10];
  const float* m2_in_w   = (const float*)d_in[11];
  const float* m2_conv_w = (const float*)d_in[12];
  const float* m2_conv_b = (const float*)d_in[13];
  const float* m2_x_w    = (const float*)d_in[14];
  const float* m2_dt_w   = (const float*)d_in[15];
  const float* m2_dt_b   = (const float*)d_in[16];
  const float* m2_A_log  = (const float*)d_in[17];
  const float* m2_D      = (const float*)d_in[18];
  const float* m2_out_w  = (const float*)d_in[19];

  uint8_t* ws = (uint8_t*)d_ws;
  size_t off = 0;
  auto alloc = [&](size_t bytes) -> void* {
    void* p = ws + off;
    off = (off + bytes + 255) & ~(size_t)255;
    return p;
  };
  _Float16* fopH  = (_Float16*)alloc((size_t)M1 * DMODEL * 2);  // 4 MB
  _Float16* inwH  = (_Float16*)alloc(512 * 128 * 2);
  _Float16* xwH   = (_Float16*)alloc(48 * 256 * 2);
  _Float16* outwH = (_Float16*)alloc(128 * 256 * 2);
  float*    xzF   = (float*)   alloc((size_t)M1 * 512 * 4);     // 32 MB (x | z)
  float*    xcF   = (float*)   alloc((size_t)M1 * 256 * 4);     // 16 MB
  _Float16* xcH   = (_Float16*)alloc((size_t)M1 * 256 * 2);     // 8 MB
  float*    dbl   = (float*)   alloc((size_t)M1 * 48 * 4);      // 3 MB
  float*    hloc  = (float*)   alloc((size_t)Bn * NCH * 256 * 16 * 4);
  float*    Pb    = (float*)   alloc((size_t)Bn * NCH * 256 * 16 * 4);
  _Float16* yactH = (_Float16*)alloc((size_t)M1 * 256 * 2);     // 8 MB
  // xzF is dead after k_scan_emit -> reuse its 32 MB for the tail buffers
  float* out1  = xzF;                       // 16384*128 floats (8 MB)
  float* resid = xzF + (size_t)M1 * 128;    // 524288 floats (2 MB)
  float* foc   = resid + (size_t)Bn * Cn * HW;
  float* sbuf  = foc + 128;

  dim3 b256(256);
  k_prep_fop<<<(M1 * DMODEL + 255) / 256, b256, 0, stream>>>(fo1, fopH);
  k_prep_w<<<(512 * 128 + 48 * 256 + 128 * 256 + 255) / 256, b256, 0, stream>>>(
      m1_in_w, m1_x_w, m1_out_w, inwH, xwH, outwH);
  // GEMM1: xz = fop @ in_w.T   (M=16384, N=512, K=128)
  k_gemm<128><<<dim3(M1 / 16, 4), dim3(256), 0, stream>>>(fopH, inwH, xzF, 512);
  k_conv<<<(M1 * DIN + 255) / 256, b256, 0, stream>>>(xzF, m1_conv_w, m1_conv_b, xcF, xcH);
  // GEMM2: dbl = xc @ x_w.T    (M=16384, N=48(pad of 40), K=256)
  k_gemm<256><<<dim3(M1 / 16, 1), dim3(96), 0, stream>>>(xcH, xwH, dbl, 48);
  // selective scan: chunked 2-pass
  k_scan_part<<<Bn * 8 * NCH, dim3(32), 0, stream>>>(dbl, xcF, m1_dt_w, m1_dt_b,
                                                     m1_A_log, hloc, Pb);
  k_scan_fix<<<Bn * 8, dim3(32), 0, stream>>>(hloc, Pb);
  k_scan_emit<<<Bn * 8 * NCH, dim3(32), 0, stream>>>(dbl, xcF, xzF, m1_dt_w, m1_dt_b,
                                                     m1_A_log, m1_D, hloc, yactH);
  // GEMM3: out1 = yact @ out_w.T (M=16384, N=128, K=256)
  k_gemm<256><<<dim3(M1 / 16, 1), dim3(256), 0, stream>>>(yactH, outwH, out1, 128);
  k_resid<<<Bn * Cn, dim3(256), 0, stream>>>(out1, fo2, resid, foc);
  k_mamba2<<<Bn, dim3(32), 0, stream>>>(foc, m2_in_w, m2_conv_w, m2_conv_b, m2_x_w,
                                        m2_dt_w, m2_dt_b, m2_A_log, m2_D, m2_out_w, sbuf);
  k_final<<<(Bn * Cn * HW + 255) / 256, b256, 0, stream>>>(resid, sbuf, (float*)d_out);
}
